// SSMLayer_81415400063622
// MI455X (gfx1250) — matrix operand order repaired and checked by reading, not compiled, not run on hardware
//
#include <hip/hip_runtime.h>
#include <hip/hip_bf16.h>

// ---------------------------------------------------------------------------
// SSM layer for MI455X (gfx1250, wave32, WMMA).
// FFT conv rewritten as exact linear recurrence y[t] = exp(A)*y[t-1] + u[t].
// All GEMMs run through v_wmma_f32_16x16x32_bf16 with f32 accumulation.
// Double-buffered LDS pipeline: one barrier per K-step, global loads for the
// next stage overlap the WMMA compute phase; global_prefetch warms stage+2.
// ---------------------------------------------------------------------------

typedef __bf16 bf16_t;
typedef __attribute__((ext_vector_type(16))) __bf16 v16bf;
typedef __attribute__((ext_vector_type(8)))  __bf16 v8bf;
typedef __attribute__((ext_vector_type(8)))  float  v8f;

union Vu { v16bf v; v8bf h[2]; };

#define D_MODEL 1024
#define D_STATE 128
#define SEQ_LEN 4096
#define BATCH   4
#define MTOT    (BATCH * SEQ_LEN)   // 16384
#define LN_EPS  1e-3f

// ---------------------------------------------------------------------------
// Weight convert + transpose: W [K][N] f32  ->  Wt [N][K] bf16
// ---------------------------------------------------------------------------
__global__ void cvt_transpose_f32_bf16(const float* __restrict__ W,
                                       bf16_t* __restrict__ Wt,
                                       int K, int N) {
    int idx = blockIdx.x * 256 + threadIdx.x;
    if (idx >= K * N) return;
    int n = idx / K;
    int k = idx - n * K;
    Wt[idx] = (bf16_t)W[(size_t)k * N + n];
}

// Fused bias for [x_state | B_sel | C_sel] projection: [0,0..,b_B,..,b_C..]
__global__ void build_bcat(const float* __restrict__ bB,
                           const float* __restrict__ bC,
                           float* __restrict__ bcat) {
    int i = threadIdx.x;            // 384 threads
    float v = 0.f;
    if (i >= 256)      v = bC[i - 256];
    else if (i >= 128) v = bB[i - 128];
    bcat[i] = v;
}

// ---------------------------------------------------------------------------
// LayerNorm over last dim (1024), emit bf16. One block (256 thr) per row.
// ---------------------------------------------------------------------------
__global__ __launch_bounds__(256)
void layernorm_to_bf16(const float* __restrict__ x,
                       const float* __restrict__ gamma,
                       const float* __restrict__ beta,
                       bf16_t* __restrict__ out) {
    const int row = blockIdx.x;
    const float* xr = x + (size_t)row * D_MODEL;
    bf16_t* orow = out + (size_t)row * D_MODEL;

    float s = 0.f, s2 = 0.f;
    for (int i = threadIdx.x; i < D_MODEL; i += 256) {
        float v = xr[i];
        s += v; s2 += v * v;
    }
    __shared__ float red[16];
    for (int off = 16; off > 0; off >>= 1) {
        s  += __shfl_down(s,  off, 32);
        s2 += __shfl_down(s2, off, 32);
    }
    const int wave = threadIdx.x >> 5, lane = threadIdx.x & 31;
    if (lane == 0) { red[wave] = s; red[8 + wave] = s2; }
    __syncthreads();
    if (wave == 0) {
        s  = (lane < 8) ? red[lane] : 0.f;
        s2 = (lane < 8) ? red[8 + lane] : 0.f;
        for (int off = 4; off > 0; off >>= 1) {
            s  += __shfl_down(s,  off, 32);
            s2 += __shfl_down(s2, off, 32);
        }
        if (lane == 0) { red[0] = s; red[8] = s2; }
    }
    __syncthreads();
    const float mu  = red[0] * (1.f / D_MODEL);
    const float var = red[8] * (1.f / D_MODEL) - mu * mu;
    const float inv = rsqrtf(var + LN_EPS);
    for (int i = threadIdx.x; i < D_MODEL; i += 256) {
        float v = (xr[i] - mu) * inv * gamma[i] + beta[i];
        orow[i] = (bf16_t)v;
    }
}

// ---------------------------------------------------------------------------
// SSM recurrence (exact rewrite of the FFT convolution):
//   u[t]  = x_state[t] * B_sel[t]
//   y[t]  = exp(A_s) * y[t-1] + u[t]
//   ys[t] = y[t] * C_sel[t]            (stored bf16 for the W_so GEMM)
// proj layout: [M][384] = [x_state | B_sel | C_sel]
// ---------------------------------------------------------------------------
__global__ __launch_bounds__(128)
void ssm_scan(const float* __restrict__ proj,
              const float* __restrict__ A_log,
              bf16_t* __restrict__ ys) {
    const int b = blockIdx.x;
    const int s = threadIdx.x;
    const float a = __expf(-__expf(A_log[s]));   // decay per step
    float y = 0.f;
    const float* pb = proj + (size_t)b * SEQ_LEN * 384;
    bf16_t* yb = ys + (size_t)b * SEQ_LEN * D_STATE;
    for (int t = 0; t < SEQ_LEN; ++t) {
        const float* pr = pb + (size_t)t * 384;
        float u = pr[s] * pr[128 + s];
        y = fmaf(a, y, u);
        yb[(size_t)t * D_STATE + s] = (bf16_t)(y * pr[256 + s]);
    }
}

// ---------------------------------------------------------------------------
// WMMA GEMM: D = A[M][K] (bf16, row-major) x Bt[N][K]^T (bf16) + epilogue.
// Block tile 128x128, BK=32, 256 threads = 8 waves arranged 4(M) x 2(N),
// each wave owns a 32x64 tile = 2x4 wmma accumulators.
// Double-buffered LDS: one barrier per K-step, next-stage global loads in
// registers overlap the WMMA phase, stage+2 warmed via global_prefetch.
// MODE 0: outB = bf16(acc + bias[n])                         (z)
// MODE 1: outF = acc + bias[n]                               (projections)
// MODE 2: outB = bf16(acc + dvec[n] * auxB[m][n])            (W_so + D*z)
// MODE 3: outF = acc + bias[n] + auxF[m][n]                  (W_out + resid)
// ---------------------------------------------------------------------------
template <int MODE>
__global__ __launch_bounds__(256)
void wmma_gemm_bf16(int M, int N, int K,
                    const bf16_t* __restrict__ A,
                    const bf16_t* __restrict__ Bt,
                    const float*  __restrict__ bias,
                    const float*  __restrict__ dvec,
                    const bf16_t* __restrict__ auxB,
                    const float*  __restrict__ auxF,
                    float*  __restrict__ outF,
                    bf16_t* __restrict__ outB) {
    constexpr int BM = 128, BN = 128, BK = 32, BKP = 40;  // pad vs bank conflicts
    __shared__ __align__(16) bf16_t As[2][BM * BKP];
    __shared__ __align__(16) bf16_t Bs[2][BN * BKP];

    const int tid   = threadIdx.x;
    const int wave  = tid >> 5;
    const int lane  = tid & 31;
    const int lr    = lane & 15;   // row/col within 16
    const int lh    = lane >> 4;   // half-wave select
    const int waveM = wave & 3;    // 0..3
    const int waveN = wave >> 2;   // 0..1
    const int bm    = blockIdx.y * BM;
    const int bn    = blockIdx.x * BN;

    const int ldr = tid >> 1;          // 0..127 : tile row for coop loads
    const int ldc = (tid & 1) * 16;    // 0 / 16 : 32B chunk within BK

    const bf16_t* gA = A  + (size_t)(bm + ldr) * K + ldc;
    const bf16_t* gB = Bt + (size_t)(bn + ldr) * K + ldc;
    const int lofs = ldr * BKP + ldc;

    v8f acc[2][4] = {};

    // ---- prologue: stage 0 -> LDS buffer 0 ----
    v8bf ra0 = ((const v8bf*)gA)[0];
    v8bf ra1 = ((const v8bf*)gA)[1];
    v8bf rb0 = ((const v8bf*)gB)[0];
    v8bf rb1 = ((const v8bf*)gB)[1];
    *(v8bf*)&As[0][lofs]     = ra0;
    *(v8bf*)&As[0][lofs + 8] = ra1;
    *(v8bf*)&Bs[0][lofs]     = rb0;
    *(v8bf*)&Bs[0][lofs + 8] = rb1;
    __syncthreads();

    const int nsteps = K / BK;
    int cur = 0;
    for (int s = 0; s < nsteps; ++s) {
        // warm L2/WGP$ for stage+2
        if (s + 2 < nsteps) {
            __builtin_prefetch(gA + (size_t)(s + 2) * BK, 0, 1);
            __builtin_prefetch(gB + (size_t)(s + 2) * BK, 0, 1);
        }
        // issue next-stage global loads (overlap with WMMA phase below)
        if (s + 1 < nsteps) {
            const bf16_t* pA = gA + (size_t)(s + 1) * BK;
            const bf16_t* pB = gB + (size_t)(s + 1) * BK;
            ra0 = ((const v8bf*)pA)[0];
            ra1 = ((const v8bf*)pA)[1];
            rb0 = ((const v8bf*)pB)[0];
            rb1 = ((const v8bf*)pB)[1];
        }

        // --- fragment gathers per documented gfx1250 VGPR layouts ---
        const bf16_t* Ac = As[cur];
        const bf16_t* Bc = Bs[cur];
        v16bf afrag[2], bfrag[4];
#pragma unroll
        for (int mt = 0; mt < 2; ++mt) {
            const int row = waveM * 32 + mt * 16 + lr;
            Vu a;
            a.h[0] = *(const v8bf*)&Ac[row * BKP + lh * 8];        // K 0-7 / 8-15
            a.h[1] = *(const v8bf*)&Ac[row * BKP + 16 + lh * 8];   // K 16-23 / 24-31
            afrag[mt] = a.v;
        }
#pragma unroll
        for (int nt = 0; nt < 4; ++nt) {
            const int col = waveN * 64 + nt * 16 + lr;
            Vu b;
            b.h[0] = *(const v8bf*)&Bc[col * BKP + lh * 8];        // K 0-7 / 8-15
            b.h[1] = *(const v8bf*)&Bc[col * BKP + 16 + lh * 8];   // K 16-23 / 24-31
            bfrag[nt] = b.v;
        }
#pragma unroll
        for (int mt = 0; mt < 2; ++mt)
#pragma unroll
            for (int nt = 0; nt < 4; ++nt)
                acc[mt][nt] = __builtin_amdgcn_wmma_f32_16x16x32_bf16(
                    false, afrag[mt], false, bfrag[nt],
                    (short)0, acc[mt][nt], false, false);

        // --- stage next buffer; single barrier per K-step ---
        if (s + 1 < nsteps) {
            const int nxt = cur ^ 1;
            *(v8bf*)&As[nxt][lofs]     = ra0;
            *(v8bf*)&As[nxt][lofs + 8] = ra1;
            *(v8bf*)&Bs[nxt][lofs]     = rb0;
            *(v8bf*)&Bs[nxt][lofs + 8] = rb1;
            __syncthreads();
            cur = nxt;
        }
    }

    // --- epilogue: lane L -> n = tile_n + L%16, VGPR r -> m = r + 8*(L/16) ---
#pragma unroll
    for (int mt = 0; mt < 2; ++mt) {
#pragma unroll
        for (int nt = 0; nt < 4; ++nt) {
            const int n = bn + waveN * 64 + nt * 16 + lr;
            const float bv = (MODE == 0 || MODE == 1 || MODE == 3) ? bias[n] : 0.f;
            const float dv = (MODE == 2) ? dvec[n] : 0.f;
#pragma unroll
            for (int r = 0; r < 8; ++r) {
                const int m = bm + waveM * 32 + mt * 16 + lh * 8 + r;
                const size_t idx = (size_t)m * N + n;
                const float v = acc[mt][nt][r];
                if (MODE == 0) {
                    outB[idx] = (bf16_t)(v + bv);
                } else if (MODE == 1) {
                    outF[idx] = v + bv;
                } else if (MODE == 2) {
                    outB[idx] = (bf16_t)(v + dv * (float)auxB[idx]);
                } else {
                    outF[idx] = v + bv + auxF[idx];
                }
            }
        }
    }
}

// ---------------------------------------------------------------------------
extern "C" void kernel_launch(void* const* d_in, const int* in_sizes, int n_in,
                              void* d_out, int out_size, void* d_ws, size_t ws_size,
                              hipStream_t stream) {
    (void)in_sizes; (void)n_in; (void)out_size; (void)ws_size;

    const float* x      = (const float*)d_in[0];
    const float* gamma  = (const float*)d_in[1];
    const float* beta   = (const float*)d_in[2];
    const float* W_in   = (const float*)d_in[3];
    const float* b_in   = (const float*)d_in[4];
    const float* W_xs   = (const float*)d_in[5];
    const float* W_B    = (const float*)d_in[6];
    const float* b_B    = (const float*)d_in[7];
    const float* W_C    = (const float*)d_in[8];
    const float* b_C    = (const float*)d_in[9];
    const float* A_log  = (const float*)d_in[10];
    const float* Dvec   = (const float*)d_in[11];
    const float* W_so   = (const float*)d_in[12];
    const float* W_out  = (const float*)d_in[13];
    const float* b_out  = (const float*)d_in[14];
    float* out = (float*)d_out;

    // -------- workspace carve-up (all intermediates ~135MB, fits L2) -------
    char* p = (char*)d_ws;
    auto carve = [&](size_t bytes) -> char* {
        char* r = p;
        p += (bytes + 255) & ~(size_t)255;
        return r;
    };
    bf16_t* xn_bf   = (bf16_t*)carve((size_t)MTOT * D_MODEL * 2);   // LN(x)
    bf16_t* z_bf    = (bf16_t*)carve((size_t)MTOT * D_MODEL * 2);   // z
    bf16_t* y_bf    = (bf16_t*)carve((size_t)MTOT * D_MODEL * 2);   // pre-W_out
    bf16_t* ys_bf   = (bf16_t*)carve((size_t)MTOT * D_STATE * 2);   // y_state*C
    float*  proj    = (float*) carve((size_t)MTOT * 384 * 4);       // xs|B|C
    bf16_t* Wt_in   = (bf16_t*)carve((size_t)D_MODEL * D_MODEL * 2);
    bf16_t* Wt_proj = (bf16_t*)carve((size_t)384 * D_MODEL * 2);
    bf16_t* Wt_so   = (bf16_t*)carve((size_t)D_MODEL * D_STATE * 2);
    bf16_t* Wt_out  = (bf16_t*)carve((size_t)D_MODEL * D_MODEL * 2);
    float*  bcat    = (float*) carve(384 * 4);

    // -------- weight prep: convert + transpose to bf16 [N][K] --------------
    {
        int n1 = D_MODEL * D_MODEL;
        cvt_transpose_f32_bf16<<<(n1 + 255) / 256, 256, 0, stream>>>(W_in, Wt_in, D_MODEL, D_MODEL);
        int n2 = D_MODEL * D_STATE;
        cvt_transpose_f32_bf16<<<(n2 + 255) / 256, 256, 0, stream>>>(W_xs, Wt_proj,                 D_MODEL, D_STATE);
        cvt_transpose_f32_bf16<<<(n2 + 255) / 256, 256, 0, stream>>>(W_B,  Wt_proj + 128 * D_MODEL, D_MODEL, D_STATE);
        cvt_transpose_f32_bf16<<<(n2 + 255) / 256, 256, 0, stream>>>(W_C,  Wt_proj + 256 * D_MODEL, D_MODEL, D_STATE);
        int n3 = D_STATE * D_MODEL;
        cvt_transpose_f32_bf16<<<(n3 + 255) / 256, 256, 0, stream>>>(W_so, Wt_so, D_STATE, D_MODEL);
        cvt_transpose_f32_bf16<<<(n1 + 255) / 256, 256, 0, stream>>>(W_out, Wt_out, D_MODEL, D_MODEL);
        build_bcat<<<1, 384, 0, stream>>>(b_B, b_C, bcat);
    }

    // -------- 1) LayerNorm -> bf16 -----------------------------------------
    layernorm_to_bf16<<<MTOT, 256, 0, stream>>>(x, gamma, beta, xn_bf);

    // -------- 2) z = xn @ W_in + b_in  (bf16 out) ---------------------------
    {
        dim3 g(D_MODEL / 128, MTOT / 128);
        wmma_gemm_bf16<0><<<g, 256, 0, stream>>>(MTOT, D_MODEL, D_MODEL,
            xn_bf, Wt_in, b_in, nullptr, nullptr, nullptr, nullptr, z_bf);
    }

    // -------- 3) [x_state | B_sel | C_sel] = z @ [W_xs|W_B|W_C] + bcat ------
    {
        dim3 g(384 / 128, MTOT / 128);
        wmma_gemm_bf16<1><<<g, 256, 0, stream>>>(MTOT, 384, D_MODEL,
            z_bf, Wt_proj, bcat, nullptr, nullptr, nullptr, proj, nullptr);
    }

    // -------- 4) SSM scan (exact recurrence replacing FFT conv) -------------
    ssm_scan<<<BATCH, D_STATE, 0, stream>>>(proj, A_log, ys_bf);

    // -------- 5) y = ys @ W_so + D*z  (bf16 out) ----------------------------
    {
        dim3 g(D_MODEL / 128, MTOT / 128);
        wmma_gemm_bf16<2><<<g, 256, 0, stream>>>(MTOT, D_MODEL, D_STATE,
            ys_bf, Wt_so, nullptr, Dvec, z_bf, nullptr, nullptr, y_bf);
    }

    // -------- 6) out = y @ W_out + b_out + residual -------------------------
    {
        dim3 g(D_MODEL / 128, MTOT / 128);
        wmma_gemm_bf16<3><<<g, 256, 0, stream>>>(MTOT, D_MODEL, D_MODEL,
            y_bf, Wt_out, b_out, nullptr, nullptr, x, out, nullptr);
    }
}
